// TritonInt4LinearV3_30855045054752
// MI455X (gfx1250) — matrix operand order repaired and checked by reading, not compiled, not run on hardware
//
#include <hip/hip_runtime.h>
#include <hip/hip_bf16.h>

// W4A16 grouped-quant GEMM, M=4 K=8192 N=28672 G=128 on MI455X (gfx1250).
//
// Roofline: weight stream = (K/2)*N int32 carriers ~= 470 MB, ~1.9 GFLOP
// -> ~4 FLOP/B, purely HBM-bound, floor ~20us at 23.3 TB/s. Strategy:
//  - v_wmma_f32_16x16x32_f16, one wave owns a 16-wide N tile; M=4 padded to 16.
//  - int4 -> f16 dequant via the 0x6400 mantissa trick: exact (nib-8) then one
//    f16 multiply by the group scale == the reference's rounding exactly.
//  - Batch a full quant group (32 NT b32 loads) before consuming for MLP.
//  - Split-K across blockIdx.y (f32 partials in d_ws + deterministic reduce)
//    to get enough waves/bytes in flight to saturate 23.3 TB/s.
//  - Addressing: uniform SGPR base (group/row) + one 32-bit per-lane VGPR
//    offset so loads lower to saddr+voffset instead of 64-bit VALU chains.

typedef _Float16 h16 __attribute__((ext_vector_type(16)));
typedef _Float16 h2v __attribute__((ext_vector_type(2)));
typedef float    f8v __attribute__((ext_vector_type(8)));
typedef unsigned u8v __attribute__((ext_vector_type(8)));
typedef unsigned u4v __attribute__((ext_vector_type(4)));

// Two int4 nibbles (lo=even k, hi=odd k) -> two f16, reference-exact:
// (0x6400|nib) is exactly 1024+nib; -1032 is integer-exact = nib-8 (v_pk_add);
// then a single f16 multiply by the group scale (v_pk_mul).
__device__ __forceinline__ h2v dequant_pair(unsigned p, h2v s2) {
  unsigned u = 0x64006400u | (p & 0xFu) | ((p & 0xF0u) << 12);
  h2v v = __builtin_bit_cast(h2v, u);
  const h2v k1032 = { (_Float16)1032.0f, (_Float16)1032.0f };
  return (v - k1032) * s2;
}

// A fragment (16-bit A 16x32 layout): lane m = lane&15 is the M row, h = lane>>4.
// a[0..15] = x[m][k0+16h .. +15]; rows m>=M are 0.
// xk is the wave-uniform pointer x+k0; aoff = m*K + 16h is the per-lane offset,
// so both b128 loads share one voffset and use ioffset 0 / 16.
__device__ __forceinline__ h16 load_a(const _Float16* __restrict__ xk,
                                      int aoff, int m, int M) {
  u4v lo = {}, hi = {};
  if (m < M) {
    lo = *(const u4v*)(xk + aoff);
    hi = *(const u4v*)(xk + aoff + 8);
  }
  u8v cat;
#pragma unroll
  for (int i = 0; i < 4; ++i) { cat[i] = lo[i]; cat[4 + i] = hi[i]; }
  return __builtin_bit_cast(h16, cat);
}

// B fragment (32x16 f16): lane%16 = N column, lanes 0-15 hold K=k0..k0+15,
// lanes 16-31 hold K=k0+16..+31; VGPR j = nibble pair of packed row r0+j.
__device__ __forceinline__ f8v wmma_tile(h16 a, const int* w8, h2v s2, f8v acc) {
  u8v bu;
#pragma unroll
  for (int j = 0; j < 8; ++j)
    bu[j] = __builtin_bit_cast(unsigned, dequant_pair((unsigned)w8[j], s2));
  h16 b = __builtin_bit_cast(h16, bu);
  return __builtin_amdgcn_wmma_f32_16x16x32_f16(false, a, false, b,
                                                (short)0, acc, false, false);
}

__global__ __launch_bounds__(256)
void int4_linear_wmma(const _Float16* __restrict__ x,
                      const int*      __restrict__ wp,
                      const _Float16* __restrict__ sc,
                      const _Float16* __restrict__ bias,
                      const int*      __restrict__ gptr,
                      _Float16*       __restrict__ out,
                      float*          __restrict__ partial, // null => single pass
                      int M, int K, int N, int segLen) {
  const int G    = *gptr;
  const int lane = threadIdx.x & 31;
  const int wv   = threadIdx.x >> 5;
  const int n0   = (blockIdx.x * (blockDim.x >> 5) + wv) << 4;
  if (n0 >= N) return;                      // wave-uniform
  const int seg    = blockIdx.y;
  const int kStart = seg * segLen;
  const int h    = lane >> 4;
  const int m    = lane & 15;
  const int ncol = n0 + m;

  // Per-lane 32-bit offsets (uniform parts stay scalar -> saddr form).
  const int loff = 8 * h * N + ncol;        // into packed-weight rows
  const int aoff = m * K + 16 * h;          // into x, row m

  f8v acc = {};

  if (G == 128 && (kStart & 127) == 0 && (segLen & 127) == 0) {
    // Fast path: one quant group (4 K-tiles) per iteration; 32 NT b32 weight
    // loads batched before any use (~4KB unique lines per wave in flight).
    const int gBeg = kStart >> 7, gEnd = (kStart + segLen) >> 7;
    for (int g = gBeg; g < gEnd; ++g) {
      const _Float16 s = (sc + (size_t)g * N)[ncol];
      const h2v s2 = { s, s };
      const int* wg = wp + (size_t)g * 64 * N;          // uniform base
      int w[32];
#pragma unroll
      for (int t = 0; t < 4; ++t)
#pragma unroll
        for (int j = 0; j < 8; ++j)
          w[t * 8 + j] =
              __builtin_nontemporal_load(wg + (size_t)(16 * t + j) * N + loff);
      const _Float16* xg = x + (size_t)g * 128;         // uniform base
      h16 a0 = load_a(xg +  0, aoff, m, M);
      h16 a1 = load_a(xg + 32, aoff, m, M);
      h16 a2 = load_a(xg + 64, aoff, m, M);
      h16 a3 = load_a(xg + 96, aoff, m, M);
      acc = wmma_tile(a0, w +  0, s2, acc);
      acc = wmma_tile(a1, w +  8, s2, acc);
      acc = wmma_tile(a2, w + 16, s2, acc);
      acc = wmma_tile(a3, w + 24, s2, acc);
    }
  } else {
    // Generic path: any G multiple of 32; group tracked incrementally (no
    // per-iteration division).
    int g = kStart / G;                 // one division per wave
    int nextB = (g + 1) * G;
    _Float16 s = (sc + (size_t)g * N)[ncol];
    const int kEnd = kStart + segLen;
    for (int k0 = kStart; k0 < kEnd; k0 += 32) {
      if (k0 >= nextB) { ++g; nextB += G; s = (sc + (size_t)g * N)[ncol]; }
      const h2v s2 = { s, s };
      const int* wg = wp + (size_t)(k0 >> 1) * N;       // uniform base
      int w[8];
#pragma unroll
      for (int j = 0; j < 8; ++j)
        w[j] = __builtin_nontemporal_load(wg + (size_t)j * N + loff);
      h16 a = load_a(x + k0, aoff, m, M);
      acc = wmma_tile(a, w, s2, acc);
    }
  }

  // C/D f32 16x16 layout: lanes 0-15 -> row v (VGPR v), col = lane;
  // lanes 16-31 -> row 8+v, col = lane-16.
  if (partial) {
    float* p = partial + (size_t)seg * M * N;
#pragma unroll
    for (int v = 0; v < 8; ++v) {
      const int row = h * 8 + v;
      if (row < M) p[(size_t)row * N + ncol] = acc[v];
    }
  } else {
    const _Float16 bv = bias[ncol];
#pragma unroll
    for (int v = 0; v < 8; ++v) {
      const int row = h * 8 + v;
      if (row < M)
        out[(size_t)row * N + ncol] = (_Float16)acc[v] + bv;
    }
  }
}

// Deterministic split-K reduction: fixed summation order, f32 -> f16, + bias
// in f16 (same two roundings as `out.astype(f16) + bias`).
__global__ __launch_bounds__(256)
void reduce_bias(const float* __restrict__ partial,
                 const _Float16* __restrict__ bias,
                 _Float16* __restrict__ out,
                 int M, int N, int segs) {
  const int n   = blockIdx.x * blockDim.x + threadIdx.x;
  const int row = blockIdx.y;
  if (n >= N) return;
  float sum = 0.f;
  for (int s = 0; s < segs; ++s)
    sum += partial[((size_t)s * M + row) * N + n];
  out[(size_t)row * N + n] = (_Float16)sum + bias[n];
}

extern "C" void kernel_launch(void* const* d_in, const int* in_sizes, int n_in,
                              void* d_out, int out_size, void* d_ws, size_t ws_size,
                              hipStream_t stream) {
  const _Float16* x    = (const _Float16*)d_in[0];
  const int*      wp   = (const int*)d_in[1];       // int32 carrier, 1 byte used
  const _Float16* sc   = (const _Float16*)d_in[2];
  const _Float16* bias = (const _Float16*)d_in[3];
  const int*      gptr = (const int*)d_in[4];       // scalar group_size (device)
  _Float16*       out  = (_Float16*)d_out;          // reference output: float16

  const int N = in_sizes[3];                        // bias length
  const long long Kll = (2LL * in_sizes[1]) / N;    // weight_packed = (K/2)*N
  const int K = (int)Kll;
  const int M = (int)(in_sizes[0] / Kll);           // x = M*K

  // Split-K for memory-level parallelism; partials must fit in d_ws and
  // segments must stay 32-aligned in K. G-alignment is checked in-kernel.
  int segs = 1;
  if (ws_size >= (size_t)4 * M * N * sizeof(float) && (K % (4 * 32)) == 0)
    segs = 4;
  else if (ws_size >= (size_t)2 * M * N * sizeof(float) && (K % (2 * 32)) == 0)
    segs = 2;
  const int segLen = K / segs;
  float* partial = (segs > 1) ? (float*)d_ws : nullptr;

  const int wavesPerBlock = 8;                      // 256 threads = 8 wave32
  const int colsPerBlock  = wavesPerBlock * 16;
  dim3 grid((N + colsPerBlock - 1) / colsPerBlock, segs);

  int4_linear_wmma<<<grid, 256, 0, stream>>>(x, wp, sc, bias, gptr, out,
                                             partial, M, K, N, segLen);
  if (segs > 1) {
    dim3 rgrid((N + 255) / 256, M);
    reduce_bias<<<rgrid, 256, 0, stream>>>(partial, bias, out, M, N, segs);
  }
}
